// LigerFusedLinearCrossEntropyLoss_77489799954818
// MI455X (gfx1250) — compile-verified, matrix operand order checked by reading, not run on hardware
//
#include <hip/hip_runtime.h>
#include <hip/hip_bf16.h>

typedef __attribute__((ext_vector_type(16))) __bf16 v16bf;
typedef __attribute__((ext_vector_type(8)))  __bf16 v8bf;
typedef __attribute__((ext_vector_type(8)))  float  v8f;

#define N_TOK 4096
#define HID   2048
#define VOC   32000
#define VS    10            // vocab splits (grid.y)
#define MT    64            // tokens per workgroup
#define NCH   128           // vocab columns per chunk (8 waves x 16)
#define CPS   25            // chunks per split: 32000/VS/NCH
#define KT    64            // K slab staged in LDS
#define KST   (HID/KT)      // 32 K-stages
#define LDA   72            // padded LDS row stride in bf16 elems (144B -> conflict-free operand reads)

#define SHUF16(a,b) __builtin_shufflevector(a,b,0,1,2,3,4,5,6,7,8,9,10,11,12,13,14,15)
#define LD16(arr,off) SHUF16(*(const v8bf*)((arr)+(off)), *(const v8bf*)((arr)+(off)+16))
#define WMMA_BF16(a,b,c) __builtin_amdgcn_wmma_f32_16x16x32_bf16(false,(a),false,(b),(short)0,(c),false,false)

__global__ __launch_bounds__(256) void flce_init(float* __restrict__ xt) {
  int i = blockIdx.x * 256 + threadIdx.x;
  if (i < N_TOK) xt[i] = 0.0f;
}

__global__ __launch_bounds__(256) void flce_main(
    const float* __restrict__ W, const float* __restrict__ X,
    const long long* __restrict__ tgt, const float* __restrict__ bias,
    float* __restrict__ xt, float* __restrict__ pmax, float* __restrict__ psum)
{
  // double-buffered operand slabs (~110 KB) + per-wave softmax state + targets
  __shared__ __align__(16) __bf16 sAh[2][MT * LDA];
  __shared__ __align__(16) __bf16 sAl[2][MT * LDA];
  __shared__ __align__(16) __bf16 sBh[2][NCH * LDA];
  __shared__ __align__(16) __bf16 sBl[2][NCH * LDA];
  __shared__ float stM[8 * MT];   // per-wave running max, per token
  __shared__ float stS[8 * MT];   // per-wave running sum-exp, per token
  __shared__ int   sT[MT];

  const int tid  = threadIdx.x;
  const int wave = tid >> 5;
  const int lane = tid & 31;
  const int mb   = blockIdx.x * MT;
  const int vsb  = blockIdx.y * (CPS * NCH);

  if (tid < MT) sT[tid] = (int)tgt[mb + tid];

  // per-wave online-softmax state lives in LDS (exclusive per-wave region, no barrier needed)
  for (int i = lane; i < MT; i += 32) {
    stM[wave * MT + i] = -__builtin_inff();
    stS[wave * MT + i] = 0.f;
  }

  const int half  = (lane < 16) ? 0 : 1;     // half-wave -> C rows +8 / operand K-offset +8
  const int lrow  = lane & 15;
  const int opOff = half ? 8 : 0;

  float4 rA[4];                              // register staging for next K-slab (A: 64x64 f32)
  float4 rB[8];                              // (B: 128x64 f32)

  #pragma unroll 1
  for (int c = 0; c < CPS; ++c) {
    const int vb = vsb + c * NCH;

    auto loadStage = [&](int kb) {
      #pragma unroll
      for (int j = 0; j < 4; ++j) {
        int idx = tid + 256 * j;
        rA[j] = *(const float4*)(X + (size_t)(mb + (idx >> 4)) * HID + kb + (idx & 15) * 4);
      }
      #pragma unroll
      for (int j = 0; j < 8; ++j) {
        int idx = tid + 256 * j;
        rB[j] = *(const float4*)(W + (size_t)(vb + (idx >> 4)) * HID + kb + (idx & 15) * 4);
      }
      if (kb + KT < HID) {  // L2 hint for the slab after next (global_prefetch_b8)
        __builtin_prefetch(X + (size_t)(mb + (tid >> 2)) * HID + kb + KT, 0, 1);
        __builtin_prefetch(W + (size_t)(vb + (tid >> 1)) * HID + kb + KT, 0, 1);
      }
    };

    auto storeStage = [&](int pb) {          // fp32 -> split-bf16 (hi + residual lo)
      #pragma unroll
      for (int j = 0; j < 4; ++j) {
        int idx = tid + 256 * j;
        int base = (idx >> 4) * LDA + (idx & 15) * 4;
        float fv[4] = {rA[j].x, rA[j].y, rA[j].z, rA[j].w};
        #pragma unroll
        for (int e = 0; e < 4; ++e) {
          __bf16 h = (__bf16)fv[e];
          sAh[pb][base + e] = h;
          sAl[pb][base + e] = (__bf16)(fv[e] - (float)h);
        }
      }
      #pragma unroll
      for (int j = 0; j < 8; ++j) {
        int idx = tid + 256 * j;
        int base = (idx >> 4) * LDA + (idx & 15) * 4;
        float fv[4] = {rB[j].x, rB[j].y, rB[j].z, rB[j].w};
        #pragma unroll
        for (int e = 0; e < 4; ++e) {
          __bf16 h = (__bf16)fv[e];
          sBh[pb][base + e] = h;
          sBl[pb][base + e] = (__bf16)(fv[e] - (float)h);
        }
      }
    };

    v8f acc[4];
    #pragma unroll
    for (int t = 0; t < 4; ++t) acc[t] = (v8f){0.f,0.f,0.f,0.f,0.f,0.f,0.f,0.f};

    // prologue: stage slab 0 into buffer 0 (safe: prev chunk's last reads hit buffer 1)
    loadStage(0);
    storeStage(0);

    int p = 0;
    #pragma unroll 1
    for (int ks = 0; ks < KST; ++ks) {
      __syncthreads();                       // slab p visible to all waves
      if (ks + 1 < KST) loadStage((ks + 1) * KT);   // overlap HBM with WMMA below

      // compute slab p: preload ALL operands per k-step, then 12 back-to-back WMMAs
      #pragma unroll
      for (int kk = 0; kk < KT; kk += 32) {
        const int bro = (wave * 16 + lrow) * LDA + kk + opOff;
        v16bf bh = LD16(sBh[p], bro);
        v16bf bl = LD16(sBl[p], bro);
        v16bf ah[4], al[4];
        #pragma unroll
        for (int t = 0; t < 4; ++t) {
          const int aro = (t * 16 + lrow) * LDA + kk + opOff;
          ah[t] = LD16(sAh[p], aro);
          al[t] = LD16(sAl[p], aro);
        }
        #pragma unroll
        for (int t = 0; t < 4; ++t) acc[t] = WMMA_BF16(ah[t], bh, acc[t]);  // hi*hi
        #pragma unroll
        for (int t = 0; t < 4; ++t) acc[t] = WMMA_BF16(ah[t], bl, acc[t]);  // hi*lo
        #pragma unroll
        for (int t = 0; t < 4; ++t) acc[t] = WMMA_BF16(al[t], bh, acc[t]);  // lo*hi
      }

      if (ks + 1 < KST) storeStage(p ^ 1);   // prev reads of buf p^1 finished before this barrier
      p ^= 1;
    }

    // ---- chunk epilogue: bias add, target-logit capture, online softmax merge ----
    const int   col  = vb + wave * 16 + lrow;      // this lane's global vocab column
    const float bcol = bias[col];
    #pragma unroll
    for (int t = 0; t < 4; ++t) {
      #pragma unroll
      for (int r = 0; r < 8; ++r) {
        float x = acc[t][r] + bcol;
        const int token = t * 16 + half * 8 + r;
        if (sT[token] == col) xt[mb + token] = x;  // unique matcher per token
        // chunk max over 16 columns (butterfly stays within the half-wave)
        float v = x;
        v = fmaxf(v, __shfl_xor(v, 8, 32));
        v = fmaxf(v, __shfl_xor(v, 4, 32));
        v = fmaxf(v, __shfl_xor(v, 2, 32));
        v = fmaxf(v, __shfl_xor(v, 1, 32));
        // chunk sum of exp(x - v)
        float e = __expf(x - v);
        e += __shfl_xor(e, 8, 32);
        e += __shfl_xor(e, 4, 32);
        e += __shfl_xor(e, 2, 32);
        e += __shfl_xor(e, 1, 32);
        if (lrow == 0) {                            // scalar merge into per-wave LDS state
          const int si = wave * MT + token;
          float mOld = stM[si];
          float mNew = fmaxf(mOld, v);
          stS[si] = stS[si] * __expf(mOld - mNew) + e * __expf(v - mNew);
          stM[si] = mNew;
        }
      }
    }
  }

  // ---- merge the 8 waves' partials for the 64 tokens, write split partials ----
  __syncthreads();
  if (tid < MT) {
    float M = -__builtin_inff();
    #pragma unroll
    for (int w = 0; w < 8; ++w) M = fmaxf(M, stM[w * MT + tid]);
    float S = 0.f;
    #pragma unroll
    for (int w = 0; w < 8; ++w) S += stS[w * MT + tid] * __expf(stM[w * MT + tid] - M);
    pmax[(size_t)(mb + tid) * VS + blockIdx.y] = M;
    psum[(size_t)(mb + tid) * VS + blockIdx.y] = S;
  }
}

__global__ __launch_bounds__(256) void flce_reduce(
    const float* __restrict__ pmax, const float* __restrict__ psum,
    const float* __restrict__ xt, const long long* __restrict__ tgt,
    float* __restrict__ out)
{
  __shared__ float sL[256];
  __shared__ int   sC[256];
  const int tid = threadIdx.x;
  float lsum = 0.f;
  int   lcnt = 0;
  for (int n = tid; n < N_TOK; n += 256) {      // fixed order -> deterministic
    float M = -__builtin_inff();
    #pragma unroll
    for (int v = 0; v < VS; ++v) M = fmaxf(M, pmax[(size_t)n * VS + v]);
    float S = 0.f;
    #pragma unroll
    for (int v = 0; v < VS; ++v) S += psum[(size_t)n * VS + v] * __expf(pmax[(size_t)n * VS + v] - M);
    float lse = M + __logf(S);
    long long tg = tgt[n];
    if (tg != -100) { lsum += lse - xt[n]; lcnt += 1; }
  }
  sL[tid] = lsum; sC[tid] = lcnt;
  __syncthreads();
  for (int s = 128; s > 0; s >>= 1) {
    if (tid < s) { sL[tid] += sL[tid + s]; sC[tid] += sC[tid + s]; }
    __syncthreads();
  }
  if (tid == 0) out[0] = sL[0] / (float)(sC[0] > 1 ? sC[0] : 1);
}

extern "C" void kernel_launch(void* const* d_in, const int* in_sizes, int n_in,
                              void* d_out, int out_size, void* d_ws, size_t ws_size,
                              hipStream_t stream) {
  (void)in_sizes; (void)n_in; (void)out_size; (void)ws_size;
  const float*     W = (const float*)d_in[0];      // [VOC, HID]
  const float*     X = (const float*)d_in[1];      // [N_TOK, HID]
  const long long* T = (const long long*)d_in[2];  // [N_TOK] int64
  const float*     B = (const float*)d_in[3];      // [VOC]

  float* ws   = (float*)d_ws;
  float* xt   = ws;                                // [N_TOK] target logits
  float* pmax = ws + N_TOK;                        // [N_TOK, VS]
  float* psum = pmax + (size_t)N_TOK * VS;         // [N_TOK, VS]  (total ~344 KB)

  flce_init  <<<dim3((N_TOK + 255) / 256), dim3(256), 0, stream>>>(xt);
  flce_main  <<<dim3(N_TOK / MT, VS),      dim3(256), 0, stream>>>(W, X, T, B, xt, pmax, psum);
  flce_reduce<<<dim3(1),                   dim3(256), 0, stream>>>(pmax, psum, xt, T, (float*)d_out);
}
